// TimeAware_LVCBlock_74603581931728
// MI455X (gfx1250) — matrix operand order repaired and checked by reading, not compiled, not run on hardware
//
#include <hip/hip_runtime.h>
#include <hip/hip_bf16.h>

#define B_      4
#define C_      32
#define COND_   80
#define LMEL    2048
#define HOP_    8
#define UP_     8
#define LAYERS_ 4
#define HID_    64
#define EMB_    512
#define LAUD    (LMEL * UP_)
#define KP_K    192      // kp_k GEMM reduction dim (64 ch * k3)
#define KP_M    6144     // kp_k GEMM rows per layer slice

typedef __attribute__((ext_vector_type(16))) _Float16 v16h;
typedef __attribute__((ext_vector_type(8)))  _Float16 v8h;
typedef __attribute__((ext_vector_type(8)))  float    v8f;
typedef _Float16 h16;

// ---------------------------------------------------------------------------
// WMMA fragment loads (CDNA5 wave32 layouts, CDNA5 ISA §7.12.2)
// A 16x32 f16: lane m = lane&15, hsel = lane>>4; elements e0..7 -> k = hsel*8+e,
//              e8..15 -> k = 16 + hsel*8 + (e-8).  Row-major LDS [m][k].
// B 32x16 f16: stored transposed in LDS as [n][k]; lane n = lane&15; element e
//              -> same k order as A (e0..7 -> hsel*8+e, e8..15 -> 16+hsel*8+(e-8)).
// ---------------------------------------------------------------------------
__device__ __forceinline__ v16h frag_a(const h16* __restrict__ p, int hsel) {
  v16h f;
#pragma unroll
  for (int e = 0; e < 8; ++e) f[e] = p[hsel * 8 + e];
#pragma unroll
  for (int e = 0; e < 8; ++e) f[8 + e] = p[16 + hsel * 8 + e];
  return f;
}
__device__ __forceinline__ v16h frag_b(const h16* __restrict__ p, int hsel) {
  v16h f;
#pragma unroll
  for (int e = 0; e < 16; ++e) f[e] = p[(e < 8) ? (hsel * 8 + e) : (16 + hsel * 8 + (e - 8))];
  return f;
}

__device__ __forceinline__ void async_b128(unsigned int lds_off,
                                           const h16* gaddr) {
  unsigned long long ga = (unsigned long long)(uintptr_t)gaddr;
  asm volatile("global_load_async_to_lds_b128 %0, %1, off"
               :: "v"(lds_off), "v"(ga) : "memory");
}

// ---------------------------------------------------------------------------
// Specialized, fully-async, double-buffered GEMM for the dominant kp_k conv:
//   outT[b][n][m] = W[m][k] * XcolT[b][n][k] + bias[m]   (f16 out, f32 acc)
// M=6144, N=2048, K=192.  6 K-chunks of 32, one-chunk DMA lookahead.
// Both tiles are contiguous -> staged entirely with GLOBAL_LOAD_ASYNC_TO_LDS.
// ---------------------------------------------------------------------------
#define LDK2 40
__global__ __launch_bounds__(256) void gemm_kpk_async(
    const h16* __restrict__ W,      // [6144][192] (layer slice)
    const h16* __restrict__ XcolT,  // [B][2048][192]
    const float* __restrict__ bias, // [6144]     (layer slice)
    h16* __restrict__ outT) {       // [B][2048][6144]
  __shared__ h16 lA[2][128 * LDK2];
  __shared__ h16 lB[2][128 * LDK2];
  const int tid = threadIdx.x;
  const int lane = tid & 31;
  const int wave = tid >> 5;
  const int wm = (wave & 3) * 32, wn = (wave >> 2) * 64;
  const int m0 = blockIdx.y * 128, n0 = blockIdx.x * 128, b = blockIdx.z;
  const h16* Abase = W + (size_t)m0 * KP_K;
  const h16* Bbase = XcolT + ((size_t)b * LMEL + n0) * KP_K;
  v8f acc[2][4] = {};

  auto issue = [&](int kc, int buf) {
#pragma unroll
    for (int it = 0; it < 2; ++it) {
      int cidx = tid + it * 256;  // 0..511 : 16B chunk -> r = c>>2, seg = c&3
      int r = cidx >> 2, seg = (cidx & 3) * 8;
      async_b128((unsigned int)(uintptr_t)&lA[buf][r * LDK2 + seg],
                 Abase + (size_t)r * KP_K + kc + seg);
      async_b128((unsigned int)(uintptr_t)&lB[buf][r * LDK2 + seg],
                 Bbase + (size_t)r * KP_K + kc + seg);
    }
  };

  issue(0, 0);
  const int row = lane & 15, hsel = lane >> 4;
#pragma unroll
  for (int c = 0; c < 6; ++c) {
    asm volatile("s_wait_asynccnt 0x0" ::: "memory");  // chunk c DMA complete
    __syncthreads();  // data visible to all; all done reading buf (c+1)&1
    if (c + 1 < 6) issue((c + 1) * 32, (c + 1) & 1);   // overlap with compute
    const h16* bufA = lA[c & 1];
    const h16* bufB = lB[c & 1];
    v16h af[2], bf[4];
#pragma unroll
    for (int mt = 0; mt < 2; ++mt)
      af[mt] = frag_a(&bufA[(wm + mt * 16 + row) * LDK2], hsel);
#pragma unroll
    for (int nt = 0; nt < 4; ++nt)
      bf[nt] = frag_b(&bufB[(wn + nt * 16 + row) * LDK2], hsel);
#pragma unroll
    for (int mt = 0; mt < 2; ++mt)
#pragma unroll
      for (int nt = 0; nt < 4; ++nt)
        acc[mt][nt] = __builtin_amdgcn_wmma_f32_16x16x32_f16(
            false, af[mt], false, bf[nt], (short)0, acc[mt][nt], false, false);
  }
  // transposed f16 store: 8 consecutive m per lane -> one aligned 16B store
#pragma unroll
  for (int mt = 0; mt < 2; ++mt)
#pragma unroll
    for (int nt = 0; nt < 4; ++nt) {
      const int nn = n0 + wn + nt * 16 + (lane & 15);
      const int mb = m0 + wm + mt * 16 + 8 * hsel;
      v8h st;
#pragma unroll
      for (int r = 0; r < 8; ++r) st[r] = (h16)(acc[mt][nt][r] + bias[mb + r]);
      *(v8h*)(&outT[((size_t)b * LMEL + nn) * KP_M + mb]) = st;
    }
}

// ---------------------------------------------------------------------------
// Generic 1-D conv as WMMA GEMM (small convs): im2col on the fly.
// A-tile async-DMA'd when fully in-bounds.
// ---------------------------------------------------------------------------
#define LDK 72
template <int KW>
__global__ __launch_bounds__(256) void conv_gemm_wmma(
    const h16* __restrict__ W, const h16* __restrict__ X,
    const float* __restrict__ bias, float* __restrict__ outF,
    h16* __restrict__ outH, const float* __restrict__ resid,
    int M, int N, int Cin, int pad, int dil, float slope) {
  __shared__ h16 lA[128 * LDK];  // [m][k]
  __shared__ h16 lB[128 * LDK];  // [n][k] (transposed im2col)
  const int tid = threadIdx.x;
  const int lane = tid & 31, wave = tid >> 5;
  const int wm = (wave & 3) * 32, wn = (wave >> 2) * 64;
  const int m0 = blockIdx.y * 128, n0 = blockIdx.x * 128, b = blockIdx.z;
  const int Ktot = Cin * KW;
  v8f acc[2][4] = {};

  for (int kc = 0; kc < Ktot; kc += 64) {
    __syncthreads();
    // ---- stage A (weights): 128 rows x 64 halfs ----
    if (m0 + 128 <= M && kc + 64 <= Ktot) {
#pragma unroll
      for (int it = 0; it < 4; ++it) {
        int cidx = tid + it * 256;  // 0..1023 chunks of 16B
        int r = cidx >> 3, seg = (cidx & 7) * 8;
        async_b128((unsigned int)(uintptr_t)&lA[r * LDK + seg],
                   W + (size_t)(m0 + r) * Ktot + kc + seg);
      }
      asm volatile("s_wait_asynccnt 0x0" ::: "memory");
    } else {
      for (int i = tid; i < 128 * 8; i += 256) {
        int r = i >> 3, seg = (i & 7) << 3;
        int gm = m0 + r;
#pragma unroll
        for (int u = 0; u < 8; ++u) {
          int k = kc + seg + u;
          lA[r * LDK + seg + u] =
              (gm < M && k < Ktot) ? W[(size_t)gm * Ktot + k] : (h16)0.0f;
        }
      }
    }
    // ---- stage B (im2col, transposed): 128 cols x 64 halfs ----
    for (int i = tid; i < 128 * 64; i += 256) {
      int nn = i >> 6, k = i & 63;
      int kk = kc + k;
      h16 v = (h16)0.0f;
      if (kk < Ktot) {
        int cin = kk / KW;  // KW is a template constant
        int j = kk - cin * KW;
        int pos = n0 + nn + dil * j - pad;
        if (pos >= 0 && pos < N) v = X[((size_t)b * Cin + cin) * N + pos];
      }
      lB[nn * LDK + k] = v;
    }
    __syncthreads();
    const int row = lane & 15, hsel = lane >> 4;
#pragma unroll
    for (int ks = 0; ks < 2; ++ks) {
      const int kb = ks * 32;
      v16h af[2], bf[4];
#pragma unroll
      for (int mt = 0; mt < 2; ++mt)
        af[mt] = frag_a(&lA[(wm + mt * 16 + row) * LDK + kb], hsel);
#pragma unroll
      for (int nt = 0; nt < 4; ++nt)
        bf[nt] = frag_b(&lB[(wn + nt * 16 + row) * LDK + kb], hsel);
#pragma unroll
      for (int mt = 0; mt < 2; ++mt)
#pragma unroll
        for (int nt = 0; nt < 4; ++nt)
          acc[mt][nt] = __builtin_amdgcn_wmma_f32_16x16x32_f16(
              false, af[mt], false, bf[nt], (short)0, acc[mt][nt], false, false);
    }
  }
  // ---- epilogue: D layout -> M = r + 8*hsel, N = lane&15 ----
  const int col = lane & 15, hsel = lane >> 4;
#pragma unroll
  for (int mt = 0; mt < 2; ++mt)
#pragma unroll
    for (int nt = 0; nt < 4; ++nt) {
      const int nn = n0 + wn + nt * 16 + col;
      const int mb = m0 + wm + mt * 16 + 8 * hsel;
#pragma unroll
      for (int r = 0; r < 8; ++r) {
        int mm = mb + r;
        if (mm < M && nn < N) {
          float v = acc[mt][nt][r];
          if (bias) v += bias[mm];
          if (slope >= 0.0f) v = (v >= 0.0f) ? v : slope * v;
          size_t o = ((size_t)b * M + mm) * N + nn;
          if (resid) v += resid[o];
          if (outH) outH[o] = (h16)v;
          else      outF[o] = v;
        }
      }
    }
}

// ---------------------------------------------------------------------------
// Small VALU kernels
// ---------------------------------------------------------------------------
__global__ void k_noise_fc(const float* __restrict__ emb,
                           const float* __restrict__ w,
                           const float* __restrict__ bias,
                           float* __restrict__ noise) {
  int i = blockIdx.x * blockDim.x + threadIdx.x;
  if (i >= B_ * COND_) return;
  int b = i / COND_, cc = i % COND_;
  float a = bias[cc];
  for (int k = 0; k < EMB_; ++k) a += emb[b * EMB_ + k] * w[cc * EMB_ + k];
  noise[i] = a;
}

__global__ void k_make_cond(const float* __restrict__ c,
                            const float* __restrict__ noise,
                            h16* __restrict__ out, int n) {
  int i = blockIdx.x * blockDim.x + threadIdx.x;
  if (i >= n) return;
  int b = i / (COND_ * LMEL);
  int cc = (i / LMEL) % COND_;
  out[i] = (h16)(c[i] + noise[b * COND_ + cc]);
}

__global__ void k_to_f16(const float* __restrict__ in, h16* __restrict__ out,
                         long n, float slope) {
  long i = (long)blockIdx.x * blockDim.x + threadIdx.x;
  if (i >= n) return;
  float v = in[i];
  if (slope >= 0.0f) v = (v >= 0.0f) ? v : slope * v;
  out[i] = (h16)v;
}

// im2col (k=3, pad=1, dil=1) of the kernel-predictor hidden state, stored
// row-contiguous in K: XcolT[b][n][c*3+j] = h[b][c][n+j-1]
__global__ void k_im2col_T(const h16* __restrict__ x, h16* __restrict__ xc) {
  int i = blockIdx.x * blockDim.x + threadIdx.x;
  const int total = B_ * LMEL * KP_K;
  if (i >= total) return;
  int k = i % KP_K;
  int n = (i / KP_K) % LMEL;
  int b = i / (KP_K * LMEL);
  int cin = k / 3, j = k - cin * 3;
  int pos = n + j - 1;
  h16 v = (h16)0.0f;
  if (pos >= 0 && pos < LMEL) v = x[((size_t)b * HID_ + cin) * LMEL + pos];
  xc[i] = v;
}

__global__ void k_add_lrelu_f16(float* __restrict__ h,
                                const float* __restrict__ add,
                                h16* __restrict__ out, long n) {
  long i = (long)blockIdx.x * blockDim.x + threadIdx.x;
  if (i >= n) return;
  float v = h[i] + add[i];
  h[i] = v;
  out[i] = (h16)((v >= 0.0f) ? v : 0.2f * v);
}

// conv_transpose1d(lrelu(x,0.2), up_w, stride=8, pad=4); k=16 -> 2 taps / output
__global__ void k_upsample(const float* __restrict__ x,
                           const float* __restrict__ w,
                           float* __restrict__ out) {
  int i = blockIdx.x * blockDim.x + threadIdx.x;
  if (i >= B_ * C_ * LAUD) return;
  int t = i % LAUD;
  int co = (i / LAUD) % C_;
  int b = i / (C_ * LAUD);
  float a = 0.0f;
  int q = (t + 4) >> 3;
  for (int s = q - 1; s <= q; ++s) {
    if (s < 0 || s >= LMEL) continue;
    int j = t + 4 - 8 * s;
    if (j < 0 || j > 15) continue;
    for (int ci = 0; ci < C_; ++ci) {
      float xv = x[((size_t)b * C_ + ci) * LMEL + s];
      xv = (xv >= 0.0f) ? xv : 0.2f * xv;
      a += xv * w[(ci * C_ + co) * 16 + j];
    }
  }
  out[i] = a;
}

// Location-variable conv + gated residual: one block per (frame t, batch b).
// kern is stored TRANSPOSED [b][t][m'] (m' = (ci*64+co2)*3+kk): the 12KB
// per-frame kernel column is one contiguous, fully-coalesced LDS fill.
__global__ __launch_bounds__(256) void k_lvc_gate(
    const float* __restrict__ y, const h16* __restrict__ kernT,
    const float* __restrict__ biasall, float* __restrict__ h, int lay) {
  __shared__ h16 km[KP_M / 1];   // 6144
  __shared__ float yw[C_ * 10];  // window positions t*8-1 .. t*8+8
  __shared__ float bsm[64];
  const int t = blockIdx.x, b = blockIdx.y, tid = threadIdx.x;
  const h16* kcol = kernT + ((size_t)b * LMEL + t) * KP_M;
  for (int i = tid; i < KP_M; i += 256) km[i] = kcol[i];
  for (int i = tid; i < C_ * 10; i += 256) {
    int ci = i / 10, u = i % 10;
    int p = t * HOP_ + u - 1;
    yw[i] = (p >= 0 && p < LAUD) ? y[((size_t)b * C_ + ci) * LAUD + p] : 0.0f;
  }
  if (tid < 64) bsm[tid] = biasall[((size_t)b * 256 + lay * 64 + tid) * LMEL + t];
  __syncthreads();
  const int co = tid >> 3, s = tid & 7;
  float as = bsm[co], at = bsm[co + 32];
  for (int ci = 0; ci < C_; ++ci) {
#pragma unroll
    for (int kk = 0; kk < 3; ++kk) {
      float xv = yw[ci * 10 + s + kk];
      as += xv * (float)km[(ci * 64 + co) * 3 + kk];
      at += xv * (float)km[(ci * 64 + co + 32) * 3 + kk];
    }
  }
  float g = (1.0f / (1.0f + expf(-as))) * tanhf(at);
  h[((size_t)b * C_ + co) * LAUD + t * HOP_ + s] += g;
}

// ---------------------------------------------------------------------------
static inline void launch_conv(dim3 grid, hipStream_t s, const h16* W,
                               const h16* X, const float* bias, float* oF,
                               h16* oH, const float* res, int M, int N,
                               int Cin, int KW, int pad, int dil, float slope) {
  dim3 blk(256);
  if (KW == 5)
    conv_gemm_wmma<5><<<grid, blk, 0, s>>>(W, X, bias, oF, oH, res, M, N, Cin,
                                           pad, dil, slope);
  else
    conv_gemm_wmma<3><<<grid, blk, 0, s>>>(W, X, bias, oF, oH, res, M, N, Cin,
                                           pad, dil, slope);
}

extern "C" void kernel_launch(void* const* d_in, const int* in_sizes, int n_in,
                              void* d_out, int out_size, void* d_ws,
                              size_t ws_size, hipStream_t stream) {
  (void)in_sizes; (void)n_in; (void)out_size; (void)ws_size;
  const float* x          = (const float*)d_in[0];
  const float* audio_down = (const float*)d_in[1];
  const float* c          = (const float*)d_in[2];
  const float* emb        = (const float*)d_in[3];
  const float* fc_t_w     = (const float*)d_in[4];
  const float* fc_t_b     = (const float*)d_in[5];
  const float* up_w       = (const float*)d_in[6];
  const float* conv_w     = (const float*)d_in[7];
  const float* kp_in_w    = (const float*)d_in[8];
  const float* kp_in_b    = (const float*)d_in[9];
  const float* kp_res_w1  = (const float*)d_in[10];
  const float* kp_res_b1  = (const float*)d_in[11];
  const float* kp_res_w2  = (const float*)d_in[12];
  const float* kp_res_b2  = (const float*)d_in[13];
  const float* kp_k_w     = (const float*)d_in[14];
  const float* kp_k_b     = (const float*)d_in[15];
  const float* kp_b_w     = (const float*)d_in[16];
  const float* kp_b_b     = (const float*)d_in[17];
  float* h = (float*)d_out;  // main activation lives in d_out

  char* ws = (char*)d_ws;
  size_t off = 0;
  auto carve = [&](size_t bytes) -> void* {
    void* p = ws + off;
    off += (bytes + 255) & ~(size_t)255;
    return p;
  };
  float* noise  = (float*)carve((size_t)B_ * COND_ * 4);
  h16* condf    = (h16*)carve((size_t)B_ * COND_ * LMEL * 2);
  h16* w_in     = (h16*)carve((size_t)64 * 400 * 2);
  h16* w_r1     = (h16*)carve((size_t)3 * 64 * 192 * 2);
  h16* w_r2     = (h16*)carve((size_t)3 * 64 * 192 * 2);
  h16* w_k      = (h16*)carve((size_t)24576 * KP_K * 2);
  h16* w_b      = (h16*)carve((size_t)256 * KP_K * 2);
  h16* w_cv     = (h16*)carve((size_t)4 * 32 * 96 * 2);
  float* hkp    = (float*)carve((size_t)B_ * HID_ * LMEL * 4);
  h16* hkp16    = (h16*)carve((size_t)B_ * HID_ * LMEL * 2);
  float* rkp    = (float*)carve((size_t)B_ * HID_ * LMEL * 4);
  h16* rkp16    = (h16*)carve((size_t)B_ * HID_ * LMEL * 2);
  h16* xcolT    = (h16*)carve((size_t)B_ * LMEL * KP_K * 2);
  float* biasall= (float*)carve((size_t)B_ * 256 * LMEL * 4);
  h16* kern     = (h16*)carve((size_t)B_ * KP_M * LMEL * 2);  // one layer slice
  h16* y16      = (h16*)carve((size_t)B_ * C_ * LAUD * 2);
  float* y2     = (float*)carve((size_t)B_ * C_ * LAUD * 4);

  const long n_hkp = (long)B_ * HID_ * LMEL;   // 524288
  const long n_aud = (long)B_ * C_ * LAUD;     // 2097152

  // weight f32 -> f16
  k_to_f16<<<(25600 + 255) / 256, 256, 0, stream>>>(kp_in_w, w_in, 25600, -1.f);
  k_to_f16<<<(36864 + 255) / 256, 256, 0, stream>>>(kp_res_w1, w_r1, 36864, -1.f);
  k_to_f16<<<(36864 + 255) / 256, 256, 0, stream>>>(kp_res_w2, w_r2, 36864, -1.f);
  k_to_f16<<<(4718592 + 255) / 256, 256, 0, stream>>>(kp_k_w, w_k, 4718592, -1.f);
  k_to_f16<<<(49152 + 255) / 256, 256, 0, stream>>>(kp_b_w, w_b, 49152, -1.f);
  k_to_f16<<<(12288 + 255) / 256, 256, 0, stream>>>(conv_w, w_cv, 12288, -1.f);

  // condition = c + (emb @ fc_t_w^T + fc_t_b)[:, :, None]
  k_noise_fc<<<2, 256, 0, stream>>>(emb, fc_t_w, fc_t_b, noise);
  k_make_cond<<<(B_ * COND_ * LMEL + 255) / 256, 256, 0, stream>>>(
      c, noise, condf, B_ * COND_ * LMEL);

  // kernel predictor: input conv (80->64, k=5, pad=2), lrelu 0.1
  launch_conv(dim3(LMEL / 128, 1, B_), stream, w_in, condf, kp_in_b, hkp,
              nullptr, nullptr, 64, LMEL, 80, 5, 2, 1, 0.1f);
  // 3 residual blocks
  for (int i = 0; i < 3; ++i) {
    k_to_f16<<<(n_hkp + 255) / 256, 256, 0, stream>>>(hkp, hkp16, n_hkp, -1.f);
    launch_conv(dim3(LMEL / 128, 1, B_), stream, w_r1 + (size_t)i * 12288,
                hkp16, kp_res_b1 + i * 64, rkp, nullptr, nullptr, 64, LMEL, 64,
                3, 1, 1, 0.1f);
    k_to_f16<<<(n_hkp + 255) / 256, 256, 0, stream>>>(rkp, rkp16, n_hkp, -1.f);
    launch_conv(dim3(LMEL / 128, 1, B_), stream, w_r2 + (size_t)i * 12288,
                rkp16, kp_res_b2 + i * 64, hkp, nullptr, hkp, 64, LMEL, 64, 3,
                1, 1, 0.1f);
  }
  k_to_f16<<<(n_hkp + 255) / 256, 256, 0, stream>>>(hkp, hkp16, n_hkp, -1.f);
  // im2col of final hidden state for the async kp_k GEMM
  k_im2col_T<<<(B_ * LMEL * KP_K + 255) / 256, 256, 0, stream>>>(hkp16, xcolT);
  // LVC biases (64->256, k=3)
  launch_conv(dim3(LMEL / 128, 2, B_), stream, w_b, hkp16, kp_b_b, biasall,
              nullptr, nullptr, 256, LMEL, 64, 3, 1, 1, -1.f);

  // main branch: h = conv_transpose(lrelu(x,0.2))
  k_upsample<<<(B_ * C_ * LAUD + 255) / 256, 256, 0, stream>>>(x, up_w, h);

  int dil = 1;
  for (int lay = 0; lay < LAYERS_; ++lay) {
    // h += audio_down; y16 = f16(lrelu(h, 0.2))
    k_add_lrelu_f16<<<(n_aud + 255) / 256, 256, 0, stream>>>(h, audio_down,
                                                             y16, n_aud);
    // dilated conv 32->32, lrelu 0.2
    launch_conv(dim3(LAUD / 128, 1, B_), stream, w_cv + (size_t)lay * 3072,
                y16, nullptr, y2, nullptr, nullptr, 32, LAUD, 32, 3, dil, dil,
                0.2f);
    // dominant GEMM: this layer's LVC kernel slice, fully-async pipeline,
    // f16 output stored transposed [b][t][m'] for coalesced LVC consumption
    gemm_kpk_async<<<dim3(LMEL / 128, KP_M / 128, B_), 256, 0, stream>>>(
        w_k + (size_t)lay * KP_M * KP_K, xcolT, kp_k_b + lay * KP_M, kern);
    // location-variable conv + gated residual into h
    k_lvc_gate<<<dim3(LMEL, B_), 256, 0, stream>>>(y2, kern, biasall, h, lay);
    dil *= 3;
  }
}
